// MSAColumnAttention_30975304139405
// MI455X (gfx1250) — matrix operand order repaired and checked by reading, not compiled, not run on hardware
//
#include <hip/hip_runtime.h>
#include <hip/hip_fp16.h>

typedef float    v2f  __attribute__((ext_vector_type(2)));
typedef float    v8f  __attribute__((ext_vector_type(8)));
typedef _Float16 v16h __attribute__((ext_vector_type(16)));
typedef int      v4i  __attribute__((ext_vector_type(4)));

constexpr int S_ = 256, I_ = 384, C_ = 32, H_ = 8, CH_ = 4, D_ = 32;

#if __has_builtin(__builtin_amdgcn_wmma_f32_16x16x4_f32)
#define HAVE_WMMA_F32X4 1
#else
#define HAVE_WMMA_F32X4 0
#endif

#if __has_builtin(__builtin_amdgcn_global_load_async_to_lds_b128) && \
    __has_builtin(__builtin_amdgcn_s_wait_asynccnt)
#define HAVE_ASYNC_LDS 1
#else
#define HAVE_ASYNC_LDS 0
#endif

#define AS1 __attribute__((address_space(1)))
#define AS3 __attribute__((address_space(3)))

// xor-shuffle within 16-lane groups via ds_swizzle (group-of-32 mode:
// offset = and_mask[4:0] | or_mask<<5 | xor_mask<<10)
#define SWZ_XOR_F(v, XM) \
  __int_as_float(__builtin_amdgcn_ds_swizzle(__float_as_int(v), (0x1F | ((XM) << 10))))

__device__ __forceinline__ float red_max16(float v) {
  v = fmaxf(v, SWZ_XOR_F(v, 1));
  v = fmaxf(v, SWZ_XOR_F(v, 2));
  v = fmaxf(v, SWZ_XOR_F(v, 4));
  v = fmaxf(v, SWZ_XOR_F(v, 8));
  return v;
}
__device__ __forceinline__ float red_sum16(float v) {
  v += SWZ_XOR_F(v, 1);
  v += SWZ_XOR_F(v, 2);
  v += SWZ_XOR_F(v, 4);
  v += SWZ_XOR_F(v, 8);
  return v;
}

// ---------------------------------------------------------------------------
// Kernel 1: LayerNorm + Q/K/V/gate projections.
// Block = 256 threads (8 waves), each wave owns 32 tokens. token = s*I + i.
// q/k/v stored [i][h][s][ch]; gate stored [token][D].
// ---------------------------------------------------------------------------
__global__ __launch_bounds__(256) void msa_ln_qkvg(
    const float* __restrict__ m,
    const float* __restrict__ lnw, const float* __restrict__ lnb,
    const float* __restrict__ Wq, const float* __restrict__ Wk,
    const float* __restrict__ Wv, const float* __restrict__ Wg,
    const float* __restrict__ bg,
    float* __restrict__ qb, float* __restrict__ kb,
    float* __restrict__ vb, float* __restrict__ gb) {
  __shared__ float mn[8][32][33];  // padded stride to spread LDS banks
  const int w = threadIdx.x >> 5;
  const int l = threadIdx.x & 31;
  const int row = l & 15, half = l >> 4;
  const int t0 = blockIdx.x * 256 + w * 32;
  const int tok = t0 + l;

  // ---- LayerNorm (one token per lane) ----
  float x[32];
  {
    const float4* mp = (const float4*)(m + (size_t)tok * C_);
#pragma unroll
    for (int q = 0; q < 8; ++q) {
      float4 t = mp[q];
      x[4 * q + 0] = t.x; x[4 * q + 1] = t.y;
      x[4 * q + 2] = t.z; x[4 * q + 3] = t.w;
    }
  }
  float mu = 0.f;
#pragma unroll
  for (int c = 0; c < 32; ++c) mu += x[c];
  mu *= (1.f / 32.f);
  float var = 0.f;
#pragma unroll
  for (int c = 0; c < 32; ++c) { float d = x[c] - mu; var += d * d; }
  var *= (1.f / 32.f);
  const float rs = rsqrtf(var + 1e-5f);
#pragma unroll
  for (int c = 0; c < 32; ++c)
    mn[w][l][c] = (x[c] - mu) * rs * lnw[c] + lnb[c];
  __syncthreads();

  // precomputed 32-bit store offsets (enables SADDR-form global stores)
  unsigned qvOff[2][8], gOff[2][8];
#pragma unroll
  for (int tt = 0; tt < 2; ++tt)
#pragma unroll
    for (int j = 0; j < 8; ++j) {
      const unsigned tk = (unsigned)(t0 + tt * 16 + j + 8 * half);
      const unsigned ii = tk % (unsigned)I_;
      const unsigned ss = tk / (unsigned)I_;
      qvOff[tt][j] = ii * (unsigned)(H_ * S_ * CH_) + ss * (unsigned)CH_;
      gOff[tt][j] = tk * (unsigned)D_;
    }

#if HAVE_WMMA_F32X4
  // A fragments (16x4 f32 layout: lanes0-15 K=2*half.., row = M)
  v2f a[2][8];
#pragma unroll
  for (int tt = 0; tt < 2; ++tt)
#pragma unroll
    for (int kc = 0; kc < 8; ++kc) {
      const int k0 = kc * 4 + 2 * half;
      a[tt][kc].x = mn[w][tt * 16 + row][k0];
      a[tt][kc].y = mn[w][tt * 16 + row][k0 + 1];
    }
#endif

  const float* Ws[4] = {Wq, Wk, Wv, Wg};
  float* Ob[3] = {qb, kb, vb};
#pragma unroll
  for (int p = 0; p < 4; ++p) {
#pragma unroll
    for (int nt = 0; nt < 2; ++nt) {
      const int n = nt * 16 + row;  // output channel (N = lane%16)
#if HAVE_WMMA_F32X4
      v2f b[8];
#pragma unroll
      for (int kc = 0; kc < 8; ++kc) {
        const int k0 = kc * 4 + 2 * half;
        b[kc].x = Ws[p][n * C_ + k0];     // B[k][n] = W[n][k]  (x @ W.T)
        b[kc].y = Ws[p][n * C_ + k0 + 1];
      }
#endif
#pragma unroll
      for (int tt = 0; tt < 2; ++tt) {
        v8f acc = {0.f, 0.f, 0.f, 0.f, 0.f, 0.f, 0.f, 0.f};
#if HAVE_WMMA_F32X4
#pragma unroll
        for (int kc = 0; kc < 8; ++kc)
          acc = __builtin_amdgcn_wmma_f32_16x16x4_f32(
              false, a[tt][kc], false, b[kc], (short)0, acc, false, false);
#else
#pragma unroll
        for (int j = 0; j < 8; ++j) {
          float s = 0.f;
          const int M = tt * 16 + j + 8 * half;
          for (int k = 0; k < 32; ++k) s += mn[w][M][k] * Ws[p][n * C_ + k];
          acc[j] = s;
        }
#endif
        if (p < 3) {
          const unsigned lo =
              (unsigned)(n >> 2) * (unsigned)(S_ * CH_) + (unsigned)(n & 3);
          float* dst = Ob[p];
#pragma unroll
          for (int j = 0; j < 8; ++j) dst[qvOff[tt][j] + lo] = acc[j];
        } else {
          const float bb = bg[n];
#pragma unroll
          for (int j = 0; j < 8; ++j)
            gb[gOff[tt][j] + (unsigned)n] =
                1.f / (1.f + __expf(-(acc[j] + bb)));
        }
      }
    }
  }
}

// ---------------------------------------------------------------------------
// Kernel 2: column attention for one (i, h, 64-row s-tile) per block.
// 4 waves; each wave owns 16 rows of s. K/V staged to LDS with async
// LDS loads (ASYNCcnt path) when available. QK^T via f32 16x16x4 WMMA,
// softmax via ds_swizzle reductions, PV via f16 16x16x32 WMMA (P staged
// in LDS for the C/D->A layout transpose).
// ---------------------------------------------------------------------------
__global__ __launch_bounds__(128) void msa_col_attn(
    const float* __restrict__ qb, const float* __restrict__ kb,
    const float* __restrict__ vb, const unsigned char* __restrict__ msk,
    float* __restrict__ ob) {
  __shared__ float4 K4[S_];
  __shared__ float4 V4[S_];
  __shared__ float am[S_];
  __shared__ _Float16 pbuf[4][16][S_];
  float* Kf = (float*)K4;
  float* Vf = (float*)V4;

  const int w = threadIdx.x >> 5;
  const int l = threadIdx.x & 31;
  const int row = l & 15, half = l >> 4;
  const int bid = blockIdx.x;
  const int i = bid >> 5;          // 32 blocks per residue column (8 heads * 4 s-tiles)
  const int h = (bid >> 2) & 7;
  const int st = bid & 3;
  const int sw = st * 64 + w * 16; // first s-row of this wave

  const size_t base = ((size_t)i * H_ + h) * S_ * CH_;
  const float* Qp = qb + base;
  const float* Kp = kb + base;
  const float* Vp = vb + base;

#if HAVE_ASYNC_LDS
  {
    const int e = threadIdx.x;  // 256 float4 per matrix, 128 threads -> 2 each
    __builtin_amdgcn_global_load_async_to_lds_b128(
        (AS1 v4i*)(Kp + (size_t)e * 4), (AS3 v4i*)&K4[e], 0, 0);
    __builtin_amdgcn_global_load_async_to_lds_b128(
        (AS1 v4i*)(Kp + (size_t)(e + 128) * 4), (AS3 v4i*)&K4[e + 128], 0, 0);
    __builtin_amdgcn_global_load_async_to_lds_b128(
        (AS1 v4i*)(Vp + (size_t)e * 4), (AS3 v4i*)&V4[e], 0, 0);
    __builtin_amdgcn_global_load_async_to_lds_b128(
        (AS1 v4i*)(Vp + (size_t)(e + 128) * 4), (AS3 v4i*)&V4[e + 128], 0, 0);
  }
  for (int e = threadIdx.x; e < S_; e += 128)
    am[e] = msk[(size_t)e * I_ + i] ? 0.f : -1e9f;
  __builtin_amdgcn_s_wait_asynccnt(0);
  __syncthreads();
#else
  for (int e = threadIdx.x; e < S_; e += 128) {
    K4[e] = ((const float4*)Kp)[e];
    V4[e] = ((const float4*)Vp)[e];
    am[e] = msk[(size_t)e * I_ + i] ? 0.f : -1e9f;
  }
  __syncthreads();
#endif

  float rm[8];
#pragma unroll
  for (int j = 0; j < 8; ++j)
    rm[j] = msk[(size_t)(sw + j + 8 * half) * I_ + i] ? 0.f : -1e9f;

  // ---- scores: 16 rows x 256 cols, one WMMA per 16-col tile ----
  v8f sc[16];
#if HAVE_WMMA_F32X4
  v2f aq;
  aq.x = Qp[(sw + row) * CH_ + 2 * half];
  aq.y = Qp[(sw + row) * CH_ + 2 * half + 1];
#else
  float qr[8][CH_];
#pragma unroll
  for (int j = 0; j < 8; ++j)
    for (int k = 0; k < CH_; ++k)
      qr[j][k] = Qp[(sw + j + 8 * half) * CH_ + k];
#endif
#pragma unroll
  for (int t16 = 0; t16 < 16; ++t16) {
    const int t = t16 * 16 + row;  // this lane's column (N = lane%16)
    v8f acc;
#if HAVE_WMMA_F32X4
    v2f bk;
    bk.x = Kf[t * CH_ + 2 * half];
    bk.y = Kf[t * CH_ + 2 * half + 1];
    v8f z = {0.f, 0.f, 0.f, 0.f, 0.f, 0.f, 0.f, 0.f};
    acc = __builtin_amdgcn_wmma_f32_16x16x4_f32(false, aq, false, bk,
                                                (short)0, z, false, false);
#else
#pragma unroll
    for (int j = 0; j < 8; ++j) {
      float s = 0.f;
      for (int k = 0; k < CH_; ++k) s += qr[j][k] * Kf[t * CH_ + k];
      acc[j] = s;
    }
#endif
    const float cm = am[t];
#pragma unroll
    for (int j = 0; j < 8; ++j) {
      const float v = acc[j] * 0.5f;  // 1/sqrt(CH)
      sc[t16][j] = (cm + rm[j] < -0.5f) ? -1e9f : v;
    }
  }

  // ---- softmax over t (row's 16 cols live across a 16-lane group) ----
  float mx[8], sm[8];
#pragma unroll
  for (int j = 0; j < 8; ++j) {
    float v = sc[0][j];
#pragma unroll
    for (int t16 = 1; t16 < 16; ++t16) v = fmaxf(v, sc[t16][j]);
    mx[j] = red_max16(v);
  }
#pragma unroll
  for (int j = 0; j < 8; ++j) {
    float s = 0.f;
#pragma unroll
    for (int t16 = 0; t16 < 16; ++t16) {
      const float e = __expf(sc[t16][j] - mx[j]);
      sc[t16][j] = e;
      s += e;
    }
    sm[j] = 1.f / red_sum16(s);
  }
  // stage normalized P (f16), row-major [row][t]
#pragma unroll
  for (int t16 = 0; t16 < 16; ++t16) {
    const int t = t16 * 16 + row;
#pragma unroll
    for (int j = 0; j < 8; ++j)
      pbuf[w][j + 8 * half][t] = (_Float16)(sc[t16][j] * sm[j]);
  }
  __syncthreads();

  // ---- O = P(16x256) @ V(256x4), f16 WMMA over 8 K-chunks of 32 ----
  v8f oa = {0.f, 0.f, 0.f, 0.f, 0.f, 0.f, 0.f, 0.f};
#pragma unroll
  for (int kc = 0; kc < 8; ++kc) {
    v16h ap, bv;
#pragma unroll
    for (int e = 0; e < 16; ++e) {
      const int j = e >> 1, o1 = e & 1;
      // A layout (16-bit 16x32): K = (j>=4?16:0) + (j&3)*2 + half*8 + o1
      const int ka = ((j & 4) << 2) + ((j & 3) << 1) + (half << 3) + o1;
      // B layout (16-bit 32x16): same K order as A
      const int kbx = ((j & 4) << 2) + ((j & 3) << 1) + (half << 3) + o1;
      ap[e] = pbuf[w][row][kc * 32 + ka];
      bv[e] = (row < CH_) ? (_Float16)Vf[(kc * 32 + kbx) * CH_ + row]
                          : (_Float16)0.f;
    }
    oa = __builtin_amdgcn_wmma_f32_16x16x32_f16(false, ap, false, bv,
                                                (short)0, oa, false, false);
  }
  if (row < CH_) {
    const unsigned lo = (unsigned)i * (unsigned)D_ + (unsigned)(h * CH_ + row);
#pragma unroll
    for (int j = 0; j < 8; ++j)
      ob[(unsigned)(sw + j + 8 * half) * (unsigned)(I_ * D_) + lo] = oa[j];
  }
}

// ---------------------------------------------------------------------------
// Kernel 3: out = (o * gate) @ Wo.T + bo
// ---------------------------------------------------------------------------
__global__ __launch_bounds__(256) void msa_out_proj(
    const float* __restrict__ ob, const float* __restrict__ gb,
    const float* __restrict__ Wo, const float* __restrict__ bo,
    float* __restrict__ out) {
  __shared__ float og[8][32][33];
  const int w = threadIdx.x >> 5;
  const int l = threadIdx.x & 31;
  const int row = l & 15, half = l >> 4;
  const int t0 = blockIdx.x * 256 + w * 32;
  const int tok = t0 + l;
  {
    const float4* op = (const float4*)(ob + (size_t)tok * D_);
    const float4* gp = (const float4*)(gb + (size_t)tok * D_);
#pragma unroll
    for (int q = 0; q < 8; ++q) {
      float4 o4 = op[q], g4 = gp[q];
      og[w][l][4 * q + 0] = o4.x * g4.x;
      og[w][l][4 * q + 1] = o4.y * g4.y;
      og[w][l][4 * q + 2] = o4.z * g4.z;
      og[w][l][4 * q + 3] = o4.w * g4.w;
    }
  }
  __syncthreads();

#if HAVE_WMMA_F32X4
  v2f a[2][8];
#pragma unroll
  for (int tt = 0; tt < 2; ++tt)
#pragma unroll
    for (int kc = 0; kc < 8; ++kc) {
      const int k0 = kc * 4 + 2 * half;
      a[tt][kc].x = og[w][tt * 16 + row][k0];
      a[tt][kc].y = og[w][tt * 16 + row][k0 + 1];
    }
#endif
#pragma unroll
  for (int nt = 0; nt < 2; ++nt) {
    const int n = nt * 16 + row;  // output channel c
#if HAVE_WMMA_F32X4
    v2f b[8];
#pragma unroll
    for (int kc = 0; kc < 8; ++kc) {
      const int k0 = kc * 4 + 2 * half;
      b[kc].x = Wo[n * D_ + k0];  // B[k=d][n=c] = Wo[c][d]
      b[kc].y = Wo[n * D_ + k0 + 1];
    }
#endif
    const float bov = bo[n];
#pragma unroll
    for (int tt = 0; tt < 2; ++tt) {
      v8f acc = {0.f, 0.f, 0.f, 0.f, 0.f, 0.f, 0.f, 0.f};
#if HAVE_WMMA_F32X4
#pragma unroll
      for (int kc = 0; kc < 8; ++kc)
        acc = __builtin_amdgcn_wmma_f32_16x16x4_f32(
            false, a[tt][kc], false, b[kc], (short)0, acc, false, false);
#else
#pragma unroll
      for (int j = 0; j < 8; ++j) {
        float s = 0.f;
        const int M = tt * 16 + j + 8 * half;
        for (int k = 0; k < 32; ++k) s += og[w][M][k] * Wo[n * D_ + k];
        acc[j] = s;
      }
#endif
#pragma unroll
      for (int j = 0; j < 8; ++j)
        out[(unsigned)(t0 + tt * 16 + j + 8 * half) * (unsigned)C_ +
            (unsigned)n] = acc[j] + bov;
    }
  }
}

// ---------------------------------------------------------------------------
extern "C" void kernel_launch(void* const* d_in, const int* in_sizes, int n_in,
                              void* d_out, int out_size, void* d_ws,
                              size_t ws_size, hipStream_t stream) {
  (void)in_sizes; (void)n_in; (void)out_size; (void)ws_size;
  const float* m = (const float*)d_in[0];
  const unsigned char* msk = (const unsigned char*)d_in[1];  // jnp.bool_ -> u8
  const float* lnw = (const float*)d_in[2];
  const float* lnb = (const float*)d_in[3];
  const float* Wq = (const float*)d_in[4];
  const float* Wk = (const float*)d_in[5];
  const float* Wv = (const float*)d_in[6];
  const float* Wg = (const float*)d_in[7];
  const float* bg = (const float*)d_in[8];
  const float* Wo = (const float*)d_in[9];
  const float* bo = (const float*)d_in[10];
  float* out = (float*)d_out;

  const size_t N = (size_t)I_ * H_ * S_ * CH_;  // 3,145,728 floats per buffer
  float* qb = (float*)d_ws;
  float* kb = qb + N;
  float* vb = kb + N;
  float* gb = vb + N;  // [token][D], same element count
  float* ob = gb + N;  // [token][D]

  msa_ln_qkvg<<<(S_ * I_) / 256, 256, 0, stream>>>(m, lnw, lnb, Wq, Wk, Wv, Wg,
                                                   bg, qb, kb, vb, gb);
  msa_col_attn<<<I_ * H_ * 4, 128, 0, stream>>>(qb, kb, vb, msk, ob);
  msa_out_proj<<<(S_ * I_) / 256, 256, 0, stream>>>(ob, gb, Wo, bo, out);
}
